// SiameseGNN_67173288509629
// MI455X (gfx1250) — matrix operand order repaired and checked by reading, not compiled, not run on hardware
//
#include <hip/hip_runtime.h>
#include <hip/hip_bf16.h>

typedef __attribute__((ext_vector_type(16))) _Float16 v16h;
typedef __attribute__((ext_vector_type(8)))  float    v8f;

// ---------------------------------------------------------------------------
// init: deg = 1.0 (self loop), mean accumulator = 0
// ---------------------------------------------------------------------------
__global__ void init_kernel(float* __restrict__ deg, float* __restrict__ mean_acc, int n) {
    int i = blockIdx.x * blockDim.x + threadIdx.x;
    if (i < n)  deg[i] = 1.0f;
    if (i < 64) mean_acc[i] = 0.0f;
}

// ---------------------------------------------------------------------------
// degree count over edge destinations
// ---------------------------------------------------------------------------
__global__ void deg_kernel(float* __restrict__ deg, const int* __restrict__ dst, int E) {
    int e = blockIdx.x * blockDim.x + threadIdx.x;
    if (e < E) unsafeAtomicAdd(&deg[dst[e]], 1.0f);
}

// ---------------------------------------------------------------------------
// hs[m,:] = dinv[m] * (A(m,:) @ W)        (WMMA f32_16x16x32_f16)
//   TRANSFORM=false: A(m,k) = A_in[m,k]                      (layer 1: raw x)
//   TRANSFORM=true : A(m,k) = relu(dinv[m]*A_in[m,k]+bias[k]) (layer 2 input)
// Writes hs to both H_out (gather source) and ACC_out (self-loop init).
// One wave computes one 16x64 row tile: 2 K-chunks x 4 N-tiles = 8 WMMAs.
// ---------------------------------------------------------------------------
template<bool TRANSFORM>
__global__ void gemm_hs_kernel(const float* __restrict__ A_in,   // [N,64]
                               const float* __restrict__ W,      // [64,64] row-major
                               const float* __restrict__ bias,   // [64] (TRANSFORM only)
                               const float* __restrict__ deg,    // [N]
                               float* __restrict__ H_out,        // [N,64]
                               float* __restrict__ ACC_out,      // [N,64]
                               int n_rows)
{
    // W staged as f16 in the exact 32x16 B-operand lane layout:
// element j of lane L (tile nt, chunk c): K = c*32 + (j/8)*16 + (L/16)*8 + j%8, N = nt*16 + L%16
    __shared__ alignas(32) _Float16 ldsW[4096];
    const int tid = threadIdx.x;
    #pragma unroll
    for (int i = 0; i < 16; ++i) {
        int flat = tid * 16 + i;                 // 256 threads * 16 = 4096 = 64*64
        int k = flat >> 6, n = flat & 63;
        int c = k >> 5, jj = k & 31, hs = (jj >> 3) & 1, j = (jj & 7) + ((jj >> 4) << 3);
        int nt = n >> 4, L = hs * 16 + (n & 15);
        ldsW[(((c * 4 + nt) * 32 + L) * 16 + j)] = (_Float16)W[k * 64 + n];
    }
    __syncthreads();

    const int wave = tid >> 5;
    const int lane = tid & 31;
    const int m0   = (blockIdx.x * 8 + wave) * 16;
    if (m0 >= n_rows) return;                    // wave-uniform: EXEC stays all-1 for WMMA

    const int half = lane >> 4;
    const int ml   = lane & 15;
    const int arow = m0 + ml;

    const float dinv_a = TRANSFORM ? rsqrtf(deg[arow]) : 0.0f;

    v8f acc[4] = {};

    #pragma unroll
    for (int c = 0; c < 2; ++c) {
        // A-operand, 16-bit 16x32 layout: elems 0..7 -> K=c*32+half*8+j,
        //                                 elems 8..15 -> K=c*32+16+half*8+j
        v16h a;
        const int kbase = c * 32 + half * 8;
        #pragma unroll
        for (int j = 0; j < 8; ++j) {
            float v0 = A_in[(size_t)arow * 64 + kbase + j];
            float v1 = A_in[(size_t)arow * 64 + kbase + 16 + j];
            if (TRANSFORM) {
                v0 = fmaxf(fmaf(dinv_a, v0, bias[kbase + j]), 0.0f);
                v1 = fmaxf(fmaf(dinv_a, v1, bias[kbase + 16 + j]), 0.0f);
            }
            a[j]     = (_Float16)v0;
            a[j + 8] = (_Float16)v1;
        }
        #pragma unroll
        for (int nt = 0; nt < 4; ++nt) {
            v16h b = *(const v16h*)&ldsW[((c * 4 + nt) * 32 + lane) * 16];
            acc[nt] = __builtin_amdgcn_wmma_f32_16x16x32_f16(
                false, a, false, b, (short)0, acc[nt], false, false);
        }
    }

    // C/D layout: VGPR r -> row m0 + 8*half + r, col nt*16 + ml
    #pragma unroll
    for (int r = 0; r < 8; ++r) {
        const int   row = m0 + half * 8 + r;
        const float s   = rsqrtf(deg[row]);      // fold dinv[src] into hs
        #pragma unroll
        for (int nt = 0; nt < 4; ++nt) {
            const float  v   = acc[nt][r] * s;
            const size_t idx = (size_t)row * 64 + nt * 16 + ml;
            H_out[idx]   = v;
            ACC_out[idx] = v;
        }
    }
}

// ---------------------------------------------------------------------------
// acc[dst] += hs[src] : 16 lanes per edge, float4 gather + f32 atomics
// ---------------------------------------------------------------------------
__global__ void scatter_kernel(const float4* __restrict__ H4,
                               float* __restrict__ ACC,
                               const int* __restrict__ src,
                               const int* __restrict__ dst,
                               int E)
{
    int t = blockIdx.x * blockDim.x + threadIdx.x;
    int e = t >> 4;
    int l = t & 15;
    if (e >= E) return;
    int s = src[e], d = dst[e];
    float4 v  = H4[(size_t)s * 16 + l];
    float* p  = ACC + (size_t)d * 64 + l * 4;
    unsafeAtomicAdd(p + 0, v.x);
    unsafeAtomicAdd(p + 1, v.y);
    unsafeAtomicAdd(p + 2, v.z);
    unsafeAtomicAdd(p + 3, v.w);
}

// ---------------------------------------------------------------------------
// mean over nodes of relu(dinv[i]*acc[i,:] + b2) -> 64-float accumulator
// one block handles 64 rows; 256 threads = 64 cols x 4 row-groups
// ---------------------------------------------------------------------------
__global__ void mean_kernel(const float* __restrict__ ACC,
                            const float* __restrict__ deg,
                            const float* __restrict__ bias,
                            float* __restrict__ mean_acc, int n)
{
    __shared__ float red[256];
    const int col  = threadIdx.x & 63;
    const int rsub = threadIdx.x >> 6;
    const int rbase = blockIdx.x * 64;
    float sum = 0.0f;
    for (int r = rsub; r < 64; r += 4) {
        int row = rbase + r;
        if (row < n) {
            float s = rsqrtf(deg[row]);
            sum += fmaxf(fmaf(s, ACC[(size_t)row * 64 + col], bias[col]), 0.0f);
        }
    }
    red[threadIdx.x] = sum;
    __syncthreads();
    if (rsub == 0) {
        float tot = red[col] + red[64 + col] + red[128 + col] + red[192 + col];
        unsafeAtomicAdd(&mean_acc[col], tot);
    }
}

// ---------------------------------------------------------------------------
// out[o] = (mean/N) . fc_w[o,:] + fc_b[o]    (tiny: one block, 64 threads)
// ---------------------------------------------------------------------------
__global__ void fc_kernel(const float* __restrict__ mean_acc,
                          const float* __restrict__ fc_w,
                          const float* __restrict__ fc_b,
                          float* __restrict__ out, float inv_n)
{
    int o = threadIdx.x;
    float sum = 0.0f;
    #pragma unroll
    for (int k = 0; k < 64; ++k)
        sum = fmaf(mean_acc[k] * inv_n, fc_w[o * 64 + k], sum);
    out[o] = sum + fc_b[o];
}

// ---------------------------------------------------------------------------
extern "C" void kernel_launch(void* const* d_in, const int* in_sizes, int n_in,
                              void* d_out, int out_size, void* d_ws, size_t ws_size,
                              hipStream_t stream)
{
    const float* x[2]  = { (const float*)d_in[0], (const float*)d_in[1] };
    const int*   ei[2] = { (const int*)d_in[2],   (const int*)d_in[3]   };
    const float* W1  = (const float*)d_in[4];
    const float* b1  = (const float*)d_in[5];
    const float* W2  = (const float*)d_in[6];
    const float* b2  = (const float*)d_in[7];
    const float* fcw = (const float*)d_in[8];
    const float* fcb = (const float*)d_in[9];

    const int N = in_sizes[0] / 64;

    float* ws       = (float*)d_ws;
    float* mean_acc = ws;                         // 64
    float* deg      = ws + 64;                    // N
    float* buf0     = deg + N;                    // N*64  (hs / gather source)
    float* bufA     = buf0 + (size_t)N * 64;      // N*64  (acc / next-layer input)
    float* out      = (float*)d_out;

    const int gemm_blocks = ((N + 15) / 16 + 7) / 8;

    for (int g = 0; g < 2; ++g) {
        const int  E    = in_sizes[2 + g] / 2;
        const int* srcp = ei[g];
        const int* dstp = ei[g] + E;

        init_kernel<<<(N + 255) / 256, 256, 0, stream>>>(deg, mean_acc, N);
        deg_kernel <<<(E + 255) / 256, 256, 0, stream>>>(deg, dstp, E);

        // layer 1: hs1 = dinv * (x @ W1); acc1 = hs1; acc1[dst] += hs1[src]
        gemm_hs_kernel<false><<<gemm_blocks, 256, 0, stream>>>(
            x[g], W1, nullptr, deg, buf0, bufA, N);
        scatter_kernel<<<((size_t)E * 16 + 255) / 256, 256, 0, stream>>>(
            (const float4*)buf0, bufA, srcp, dstp, E);

        // layer 2: A = relu(dinv*acc1 + b1); hs2 = dinv*(A @ W2); acc2 init = hs2
        // (alias-safe: each wave fully reads its 16 rows before storing them)
        gemm_hs_kernel<true><<<gemm_blocks, 256, 0, stream>>>(
            bufA, W2, b1, deg, buf0, bufA, N);
        scatter_kernel<<<((size_t)E * 16 + 255) / 256, 256, 0, stream>>>(
            (const float4*)buf0, bufA, srcp, dstp, E);

        // mean of relu(dinv*acc2 + b2), then FC head
        mean_kernel<<<(N + 63) / 64, 256, 0, stream>>>(bufA, deg, b2, mean_acc, N);
        fc_kernel<<<1, 64, 0, stream>>>(mean_acc, fcw, fcb, out + g * 64, 1.0f / (float)N);
    }
}
